// GNNActor_34651796144179
// MI455X (gfx1250) — matrix operand order repaired and checked by reading, not compiled, not run on hardware
//
#include <hip/hip_runtime.h>
#include <hip/hip_bf16.h>
#include <math.h>

typedef __attribute__((ext_vector_type(16))) _Float16 v16h;
typedef __attribute__((ext_vector_type(8)))  _Float16 v8h;
typedef __attribute__((ext_vector_type(8)))  float    v8f;

#define N_NODES 100000
#define ACT_DIM 1000
#define NBATCH  100
#define CIN     128
#define HID     256
#define NEDGE   1600000

// ---------------------------------------------------------------------------
// Small utility kernels
// ---------------------------------------------------------------------------
__global__ void k_f32_to_f16(const float* __restrict__ in, _Float16* __restrict__ out, int n) {
    int i = blockIdx.x * blockDim.x + threadIdx.x;
    if (i < n) out[i] = (_Float16)in[i];
}

__global__ void k_fill(float* __restrict__ p, float v, int n) {
    int i = blockIdx.x * blockDim.x + threadIdx.x;
    if (i < n) p[i] = v;
}

__global__ void k_deg_edges(const int* __restrict__ dst, float* __restrict__ deg, int ne) {
    int i = blockIdx.x * blockDim.x + threadIdx.x;
    if (i < ne) atomicAdd(&deg[dst[i]], 1.0f);
}

__global__ void k_rsqrt(const float* __restrict__ deg, float* __restrict__ dinv, int n) {
    int i = blockIdx.x * blockDim.x + threadIdx.x;
    if (i < n) dinv[i] = rsqrtf(deg[i]);
}

// h *= dinv[row]   (row-major [N, 128])
__global__ void k_rowscale(float* __restrict__ h, const float* __restrict__ dinv, int total) {
    int i = blockIdx.x * blockDim.x + threadIdx.x;
    if (i < total) h[i] *= dinv[i >> 7];
}

// agg[dst] += hs[src]   (one wave handles one edge: 32 lanes x 4 floats)
__global__ void k_scatter(const float* __restrict__ hs, const int* __restrict__ src,
                          const int* __restrict__ dst, float* __restrict__ agg, int ne) {
    long tid = (long)blockIdx.x * blockDim.x + threadIdx.x;
    long e = tid >> 5;
    if (e >= ne) return;
    int q = (int)(tid & 31) * 4;
    int s = src[e], d = dst[e];
    float4 v = *(const float4*)(hs + (size_t)s * CIN + q);
    float* ap = agg + (size_t)d * CIN + q;
    atomicAdd(ap + 0, v.x);
    atomicAdd(ap + 1, v.y);
    atomicAdd(ap + 2, v.z);
    atomicAdd(ap + 3, v.w);
}

// y = relu((agg + hs) * dinv[row] + b[col]) + x   (written in place over hs)
__global__ void k_combine(float* __restrict__ hs, const float* __restrict__ agg,
                          const float* __restrict__ dinv, const float* __restrict__ b,
                          const float* __restrict__ x, int total) {
    int i = blockIdx.x * blockDim.x + threadIdx.x;
    if (i >= total) return;
    int row = i >> 7, col = i & 127;
    float v = (agg[i] + hs[i]) * dinv[row] + b[col];
    v = v > 0.0f ? v : 0.0f;
    hs[i] = v + x[i];
}

// ---------------------------------------------------------------------------
// WMMA GEMM:  C[rowBase..+128, colBase..+128] = act(A @ Bw + bias)
//   A: f32 [nrows, K] (converted to f16 while staging in LDS)
//   Bw: f16 [K, nfull] (staged transposed into LDS -> contiguous B fragments)
//   LDS rows padded by 16 halves (32 B): keeps vector-load alignment while
//   turning the per-lane bank step from 0 (16-way conflict) into 8 (2-way).
//   8 wave32 per block; wave w owns row-tile w, loops over 8 col-tiles.
//   B fragments are batched per k-step so WMMA overlaps outstanding DS loads.
// ---------------------------------------------------------------------------
template <int K>
__global__ __launch_bounds__(256) void gemm_wmma_f16(
    const float* __restrict__ A, const _Float16* __restrict__ Bw,
    float* __restrict__ C, const float* __restrict__ bias,
    int nrows, int nfull, int act) {
    constexpr int LDA = K + 16;  // padded LDS row stride (halves)
    extern __shared__ __attribute__((aligned(32))) char smem_raw[];
    _Float16* As = (_Float16*)smem_raw;                // [128][LDA]
    _Float16* Bs = ((_Float16*)smem_raw) + 128 * LDA;  // [128 cols][LDA] (transposed)

    const int tid = threadIdx.x;
    const int rowBase = blockIdx.x * 128;
    const int colBase = blockIdx.y * 128;

    // Stage A (f32 -> f16), zero-pad rows beyond nrows
    for (int idx = tid; idx < 128 * K; idx += 256) {
        int r = idx / K, k = idx - r * K;
        int gr = rowBase + r;
        float v = (gr < nrows) ? A[(size_t)gr * K + k] : 0.0f;
        As[r * LDA + k] = (_Float16)v;
    }
    // Stage B transposed: Bs[n*LDA + k] = Bw[k*nfull + colBase + n]
    for (int idx = tid; idx < 128 * K; idx += 256) {
        int k = idx >> 7, n = idx & 127;
        Bs[n * LDA + k] = Bw[(size_t)k * nfull + colBase + n];
    }
    __syncthreads();

    const int wave = tid >> 5;
    const int lane = tid & 31;
    const int hh   = lane >> 4;   // which half of the wave
    const int l16  = lane & 15;

    v8f acc[8] = {};
    const int m = wave * 16 + l16;  // A-fragment row (lane gives M)

    for (int kt = 0; kt < K / 32; ++kt) {
        const int kb = kt * 32;
        // A fragment: per-lane K = kb + hh*16 + {0..15}
        v8h alo = *(const v8h*)(As + m * LDA + kb + hh * 16);
        v8h ahi = *(const v8h*)(As + m * LDA + kb + hh * 16 + 8);
        v16h a;
#pragma unroll
        for (int j = 0; j < 8; ++j) { a[j] = alo[j]; a[j + 8] = ahi[j]; }

        // Batch all 8 B fragments first so WMMAs can overlap DS loads
        v16h bf[8];
#pragma unroll
        for (int nt = 0; nt < 8; ++nt)
            bf[nt] = *(const v16h*)(Bs + (nt * 16 + l16) * LDA + kb + hh * 16);
#pragma unroll
        for (int nt = 0; nt < 8; ++nt)
            acc[nt] = __builtin_amdgcn_wmma_f32_16x16x32_f16(
                false, a, false, bf[nt], (short)0, acc[nt], false, false);
    }

    // Store D: lane l16 -> N, VGPR r -> M = hh*8 + r
#pragma unroll
    for (int nt = 0; nt < 8; ++nt) {
        const int col = colBase + nt * 16 + l16;
        const float bv = bias ? bias[col] : 0.0f;
#pragma unroll
        for (int r = 0; r < 8; ++r) {
            int row = rowBase + wave * 16 + hh * 8 + r;
            if (row < nrows) {
                float v = acc[nt][r] + bv;
                if (act == 1) v = (v > 0.0f) ? v : 0.01f * v;
                C[(size_t)row * nfull + col] = v;
            }
        }
    }
}

// ---------------------------------------------------------------------------
// Final layer: conc = softplus(t2 @ W3 + b3) + 1e-20   (one wave per row)
// ---------------------------------------------------------------------------
__global__ void k_final_softplus(const float* __restrict__ t2, const float* __restrict__ W3,
                                 const float* __restrict__ b3, float* __restrict__ conc, int n) {
    int wid = (blockIdx.x * blockDim.x + threadIdx.x) >> 5;
    int lane = threadIdx.x & 31;
    if (wid >= n) return;
    const float4* row = (const float4*)(t2 + (size_t)wid * HID);
    const float4* w   = (const float4*)W3;
    float s = 0.0f;
#pragma unroll
    for (int j = 0; j < 2; ++j) {
        float4 v = row[lane + j * 32];
        float4 ww = w[lane + j * 32];
        s += v.x * ww.x + v.y * ww.y + v.z * ww.z + v.w * ww.w;
    }
    for (int off = 16; off > 0; off >>= 1) s += __shfl_xor(s, off, 32);
    if (lane == 0) {
        float xv = s + b3[0];
        float sp = (xv > 20.0f) ? xv : log1pf(expf(xv));
        conc[wid] = sp + 1e-20f;
    }
}

// ---------------------------------------------------------------------------
// Deterministic counter-based gamma sampler (Marsaglia-Tsang) — stateless.
// ---------------------------------------------------------------------------
__device__ inline unsigned pcg_step(unsigned& s) {
    s = s * 747796405u + 2891336453u;
    unsigned w = ((s >> ((s >> 28) + 4u)) ^ s) * 277803737u;
    return (w >> 22) ^ w;
}
__device__ inline float u01(unsigned h) {
    return (float)(h >> 8) * (1.0f / 16777216.0f) + (0.5f / 16777216.0f);
}

__global__ void k_gamma_sample(const float* __restrict__ conc, float* __restrict__ g, int n) {
    int i = blockIdx.x * blockDim.x + threadIdx.x;
    if (i >= n) return;
    float a = conc[i];
    unsigned st = 0x9E3779B9u ^ (unsigned)i;
    pcg_step(st);
    float boost = 1.0f, a1 = a;
    if (a < 1.0f) {
        float u = u01(pcg_step(st));
        boost = powf(u, 1.0f / a);
        a1 = a + 1.0f;
    }
    float d = a1 - (1.0f / 3.0f);
    float c = rsqrtf(9.0f * d);
    float gv = d;
    for (int it = 0; it < 64; ++it) {
        float u1 = u01(pcg_step(st));
        float u2 = u01(pcg_step(st));
        float u3 = u01(pcg_step(st));
        float xn = sqrtf(-2.0f * logf(u1)) * cosf(6.2831853f * u2);
        float t = 1.0f + c * xn;
        if (t <= 0.0f) continue;
        float v = t * t * t;
        if (logf(u3) < 0.5f * xn * xn + d - d * v + d * logf(v)) { gv = d * v; break; }
    }
    g[i] = fmaxf(gv * boost, 1e-30f);
}

// ---------------------------------------------------------------------------
// Per-batch Dirichlet normalize + log_prob (one block per batch row of 1000)
// ---------------------------------------------------------------------------
__global__ void k_dirichlet_finalize(const float* __restrict__ g, const float* __restrict__ conc,
                                     float* __restrict__ action, float* __restrict__ logprob) {
    int b = blockIdx.x;
    int t = threadIdx.x;
    const float* gb = g + (size_t)b * ACT_DIM;
    const float* cb = conc + (size_t)b * ACT_DIM;
    float sg = 0, sc = 0, slg = 0, sl = 0;
    for (int i = t; i < ACT_DIM; i += blockDim.x) {
        float gi = gb[i], ci = cb[i];
        sg += gi;
        sc += ci;
        slg += lgammaf(ci);
        sl += (ci - 1.0f) * logf(gi);
    }
    __shared__ float r0[256], r1[256], r2[256], r3[256];
    __shared__ float Sg;
    r0[t] = sg; r1[t] = sc; r2[t] = slg; r3[t] = sl;
    __syncthreads();
    for (int s = 128; s > 0; s >>= 1) {
        if (t < s) { r0[t] += r0[t + s]; r1[t] += r1[t + s]; r2[t] += r2[t + s]; r3[t] += r3[t + s]; }
        __syncthreads();
    }
    if (t == 0) {
        Sg = r0[0];
        float logSg = logf(r0[0]);
        logprob[b] = r3[0] - logSg * (r1[0] - (float)ACT_DIM) + lgammaf(r1[0]) - r2[0];
    }
    __syncthreads();
    float inv = 1.0f / Sg;
    for (int i = t; i < ACT_DIM; i += blockDim.x)
        action[(size_t)b * ACT_DIM + i] = gb[i] * inv;
}

// ---------------------------------------------------------------------------
// Launcher
// ---------------------------------------------------------------------------
extern "C" void kernel_launch(void* const* d_in, const int* in_sizes, int n_in,
                              void* d_out, int out_size, void* d_ws, size_t ws_size,
                              hipStream_t stream) {
    (void)in_sizes; (void)n_in; (void)out_size; (void)ws_size;
    const float* x  = (const float*)d_in[0];
    const float* Wc = (const float*)d_in[1];
    const float* bc = (const float*)d_in[2];
    const float* W1 = (const float*)d_in[3];
    const float* b1 = (const float*)d_in[4];
    const float* W2 = (const float*)d_in[5];
    const float* b2 = (const float*)d_in[6];
    const float* W3 = (const float*)d_in[7];
    const float* b3 = (const float*)d_in[8];
    const int*   ei = (const int*)d_in[9];
    const int* esrc = ei;
    const int* edst = ei + NEDGE;

    char* ws = (char*)d_ws;
    // big activation buffers (t2 overlays hbuf+agg which are dead by then)
    float* hbuf = (float*)(ws + 0);          // [N,128] f32: h -> hs -> y (in place)
    float* agg  = (float*)(ws + 51200000);   // [N,128] f32
    float* t1   = (float*)(ws + 102400000);  // [N,256] f32
    float* t2   = (float*)(ws + 0);          // [N,256] f32 (reuse)
    char* sm = ws + 204800000;
    float* deg  = (float*)(sm + 0);
    float* dinv = (float*)(sm + 400000);
    float* conc = (float*)(sm + 800000);
    float* gam  = (float*)(sm + 1200000);
    _Float16* Wc16 = (_Float16*)(sm + 1600000);            // 128x128
    _Float16* W116 = (_Float16*)(sm + 1600000 + 32768);    // 128x256
    _Float16* W216 = (_Float16*)(sm + 1600000 + 98304);    // 256x256

    const int TPB = 256;
    const int totC = N_NODES * CIN;  // 12.8M

    // 1) weights -> f16
    k_f32_to_f16<<<(CIN * CIN + TPB - 1) / TPB, TPB, 0, stream>>>(Wc, Wc16, CIN * CIN);
    k_f32_to_f16<<<(CIN * HID + TPB - 1) / TPB, TPB, 0, stream>>>(W1, W116, CIN * HID);
    k_f32_to_f16<<<(HID * HID + TPB - 1) / TPB, TPB, 0, stream>>>(W2, W216, HID * HID);

    // 2) degrees (self-loop => init 1), zero agg
    k_fill<<<(N_NODES + TPB - 1) / TPB, TPB, 0, stream>>>(deg, 1.0f, N_NODES);
    k_fill<<<(totC + TPB - 1) / TPB, TPB, 0, stream>>>(agg, 0.0f, totC);
    k_deg_edges<<<(NEDGE + TPB - 1) / TPB, TPB, 0, stream>>>(edst, deg, NEDGE);
    k_rsqrt<<<(N_NODES + TPB - 1) / TPB, TPB, 0, stream>>>(deg, dinv, N_NODES);

    // 3) h = x @ W_conv (WMMA f16)
    dim3 gA((N_NODES + 127) / 128, 1);
    gemm_wmma_f16<CIN><<<gA, TPB, 2 * 128 * (CIN + 16) * sizeof(_Float16), stream>>>(
        x, Wc16, hbuf, nullptr, N_NODES, CIN, 0);

    // 4) hs = h * dinv[row] ; agg[dst] += hs[src] ; y = relu((agg+hs)*dinv+b)+x
    k_rowscale<<<(totC + TPB - 1) / TPB, TPB, 0, stream>>>(hbuf, dinv, totC);
    long sthreads = (long)NEDGE * 32;
    k_scatter<<<(unsigned)((sthreads + TPB - 1) / TPB), TPB, 0, stream>>>(hbuf, esrc, edst, agg, NEDGE);
    k_combine<<<(totC + TPB - 1) / TPB, TPB, 0, stream>>>(hbuf, agg, dinv, bc, x, totC);

    // 5) t1 = leaky_relu(y @ W1 + b1)   [K=128 -> 256]
    dim3 gB((N_NODES + 127) / 128, HID / 128);
    gemm_wmma_f16<CIN><<<gB, TPB, 2 * 128 * (CIN + 16) * sizeof(_Float16), stream>>>(
        hbuf, W116, t1, b1, N_NODES, HID, 1);

    // 6) t2 = leaky_relu(t1 @ W2 + b2)  [K=256 -> 256]
    gemm_wmma_f16<HID><<<gB, TPB, 2 * 128 * (HID + 16) * sizeof(_Float16), stream>>>(
        t1, W216, t2, b2, N_NODES, HID, 1);

    // 7) conc = softplus(t2 @ W3 + b3) + 1e-20   (one wave per row)
    k_final_softplus<<<(N_NODES * 32 + TPB - 1) / TPB, TPB, 0, stream>>>(t2, W3, b3, conc, N_NODES);

    // 8) Dirichlet: gamma samples, then normalize + log_prob
    k_gamma_sample<<<(N_NODES + TPB - 1) / TPB, TPB, 0, stream>>>(conc, gam, N_NODES);
    float* out_action  = (float*)d_out;
    float* out_logprob = (float*)d_out + N_NODES;
    k_dirichlet_finalize<<<NBATCH, TPB, 0, stream>>>(gam, conc, out_action, out_logprob);
}
